// ODERecFWP_84859963834862
// MI455X (gfx1250) — matrix operand order repaired and checked by reading, not compiled, not run on hardware
//
#include <hip/hip_runtime.h>
#include <hip/hip_bf16.h>

// ---------------------------------------------------------------------------
// CDNA5 / gfx1250 fused ODE-recurrent fast-weight-programmer step.
// All GEMMs use V_WMMA_F32_16X16X32_BF16 (bf16 A/B, f32 accum), wave32.
// Branch-free transcendentals + B-tile prefetch; bounded unroll (no spills).
// ---------------------------------------------------------------------------

typedef __bf16 bf16_t;
typedef __bf16 v16bf __attribute__((ext_vector_type(16)));
typedef float  v8f   __attribute__((ext_vector_type(8)));

#define DEV_INLINE __device__ __forceinline__

constexpr int N_ = 256, T_ = 64, DS_ = 32, DA_ = 8;
constexpr int D_ = 512, H_ = 8, FF_ = 2048, HD_ = 64;
constexpr int KX_  = 576;   // padded IN+D (=552) for kv GEMM
constexpr int NKV_ = 1040;  // padded 2D+H (=1032)
constexpr int KIN_ = 64;    // padded IN (=40) for W_in GEMM

// workspace layout (bytes, all 16B aligned)
constexpr size_t OFF_H   = 0;                                   // h state  N*D f32
constexpr size_t OFF_W2  = OFF_H   + (size_t)N_*D_*4;           // fast wts N*H*HD*HD f32
constexpr size_t OFF_ODE = OFF_W2  + (size_t)N_*H_*HD_*HD_*4;   // 512x512 bf16 (transposed)
constexpr size_t OFF_IN  = OFF_ODE + (size_t)D_*D_*2;           // 512x64
constexpr size_t OFF_Q   = OFF_IN  + (size_t)D_*KIN_*2;         // 512x512
constexpr size_t OFF_KV  = OFF_Q   + (size_t)D_*D_*2;           // 1040x576
constexpr size_t OFF_OUT = OFF_KV  + (size_t)NKV_*KX_*2;        // 512x512
constexpr size_t OFF_FF1 = OFF_OUT + (size_t)D_*D_*2;           // 2048x512
constexpr size_t OFF_FF2 = OFF_FF1 + (size_t)FF_*D_*2;          // 512x2048
constexpr size_t OFF_DEC = OFF_FF2 + (size_t)D_*FF_*2;          // 32x512

DEV_INLINE bf16_t f2bf(float f) {
  unsigned u = __float_as_uint(f);
  u += 0x7fffu + ((u >> 16) & 1u);          // round-to-nearest-even
  unsigned short h = (unsigned short)(u >> 16);
  return *reinterpret_cast<bf16_t*>(&h);
}

// Branch-free fast transcendentals (single v_exp_f32 / v_rcp_f32 / v_rsq_f32)
DEV_INLINE float fast_tanh(float x) {
  float ax = fabsf(x);
  float e  = __expf(-2.0f * ax);
  float r  = (1.0f - e) * __builtin_amdgcn_rcpf(1.0f + e);
  return copysignf(r, x);
}
DEV_INLINE float fast_sigmoid(float x) {
  return __builtin_amdgcn_rcpf(1.0f + __expf(-x));
}

union V16 { v16bf v; uint4 q[2]; };

// One 16x16 f32 tile: A from LDS (row-major, lda bf16 elems),
// B from global transposed weight (N-major rows of K, ldb bf16 elems).
DEV_INLINE v8f wmma_k(const bf16_t* __restrict__ aLds, int lda,
                      const bf16_t* __restrict__ bGlb, int ldb,
                      int kchunks, v8f c) {
  const int lane = threadIdx.x & 31;
  const int m    = lane & 15;
  const int half = lane >> 4;
  const bf16_t* ap = aLds + m * lda + 8 * half;          // A: K {8h..8h+7, 16+8h..}
  const bf16_t* bp = bGlb + (size_t)m * ldb + 8 * half;  // B: col m, K {8h..8h+7, 16+8h..}
#pragma unroll 4
  for (int kc = 0; kc < kchunks; ++kc) {
    V16 ua, ub;
    ua.q[0] = *(const uint4*)(ap);
    ua.q[1] = *(const uint4*)(ap + 16);
    ub.q[0] = *(const uint4*)(bp);
    ub.q[1] = *(const uint4*)(bp + 16);
    c = __builtin_amdgcn_wmma_f32_16x16x32_bf16(false, ua.v, false, ub.v,
                                                (short)0, c, false, false);
    ap += 32; bp += 32;
  }
  return c;
}

// Prefetch the next B tile's per-lane row start (global_prefetch_b8 path)
DEV_INLINE void prefetchB(const bf16_t* __restrict__ bGlb, int ldb) {
  const int lane = threadIdx.x & 31;
  __builtin_prefetch(bGlb + (size_t)(lane & 15) * ldb + 16 * (lane >> 4), 0, 0);
}

// ---------------------------------------------------------------------------
// Weight transpose + f32->bf16 convert:  dst[n*Kpad + k] = src[k*Ncol + n]
// ---------------------------------------------------------------------------
__global__ void convert_wT(const float* __restrict__ src, bf16_t* __restrict__ dst,
                           int K, int Ncol, int Kpad, int Npad) {
  int idx = blockIdx.x * blockDim.x + threadIdx.x;
  int total = Kpad * Npad;
  if (idx >= total) return;
  int n = idx / Kpad, k = idx - n * Kpad;
  float v = (k < K && n < Ncol) ? src[(size_t)k * Ncol + n] : 0.0f;
  dst[(size_t)n * Kpad + k] = f2bf(v);
}

__global__ void zero_f32(float* __restrict__ p, int n) {
  int i = blockIdx.x * blockDim.x + threadIdx.x;
  if (i < n) p[i] = 0.0f;
}

// ---------------------------------------------------------------------------
// Fused per-timestep kernel. grid = N/16 workgroups, 256 threads (8 waves).
// Each workgroup owns a 16-row batch tile (M=16 == WMMA M).
// ---------------------------------------------------------------------------
__launch_bounds__(256)
__global__ void step_kernel(
    const float* __restrict__ states, const float* __restrict__ actions,
    const float* __restrict__ tsteps,
    const float* __restrict__ b_ode, const float* __restrict__ b_in,
    const float* __restrict__ ln_in_g, const float* __restrict__ ln_in_b,
    const float* __restrict__ b_q, const float* __restrict__ b_kv,
    const float* __restrict__ b_out, const float* __restrict__ ln_ff_g,
    const float* __restrict__ ln_ff_b, const float* __restrict__ b_ff1,
    const float* __restrict__ b_ff2, const float* __restrict__ b_dec,
    const bf16_t* __restrict__ wodeT, const bf16_t* __restrict__ winT,
    const bf16_t* __restrict__ wqT,   const bf16_t* __restrict__ wkvT,
    const bf16_t* __restrict__ woutT, const bf16_t* __restrict__ wff1T,
    const bf16_t* __restrict__ wff2T, const bf16_t* __restrict__ wdecT,
    float* __restrict__ hState, float* __restrict__ W2g,
    float* __restrict__ outp, int t)
{
  __shared__ __align__(16) bf16_t sA[16 * 1024];   // 32KB A staging (also f32 "out" alias)
  __shared__ __align__(16) bf16_t sY[16 * 512];    // 16KB second A buffer
  __shared__ __align__(16) float  sF[16 * 1040];   // 66.5KB activation scratch
  __shared__ __align__(16) float  sH[16 * 512];    // 32KB h / q / o tile
  __shared__ float sRed[512];
  __shared__ float sMV[32];
  __shared__ float sDT[16];

  const int tid  = threadIdx.x;
  const int lane = tid & 31;
  const int wave = tid >> 5;
  const int r0   = blockIdx.x * 16;
  const int nn   = lane & 15, half = lane >> 4;

  // ---- Phase 0: load carried h tile + per-row dt ----
  for (int i = tid; i < 16 * 512; i += 256)
    sH[i] = hState[(size_t)(r0 + (i >> 9)) * 512 + (i & 511)];
  if (tid < 16) {
    float t0 = tsteps[(size_t)(r0 + tid) * (T_ + 1) + t];
    float t1 = tsteps[(size_t)(r0 + tid) * (T_ + 1) + t + 1];
    sDT[tid] = (t1 - t0) * 0.25f;
  }
  __syncthreads();

  // ---- Phase 1: latent ODE, 4 Euler steps of h += dt/4 * tanh(h@W_ode+b) ----
  for (int e = 0; e < 4; ++e) {
    for (int i = tid; i < 16 * 512; i += 256) sA[i] = f2bf(sH[i]);
    __syncthreads();
#pragma unroll 1
    for (int tn = wave; tn < 32; tn += 8) {
      if (tn + 8 < 32) prefetchB(wodeT + (size_t)(tn + 8) * 16 * 512, 512);
      v8f cz = {};
      v8f c = wmma_k(sA, 512, wodeT + (size_t)tn * 16 * 512, 512, 16, cz);
      int col = tn * 16 + nn;
      float bb = b_ode[col];
#pragma unroll
      for (int g = 0; g < 8; ++g) {
        int row = g + 8 * half;
        sH[row * 512 + col] += sDT[row] * fast_tanh(c[g] + bb);
      }
    }
    __syncthreads();
  }

  // ---- Phase 2: x = [data, h1] (K padded to 576); kvb = x @ W_kv + b_kv ----
  for (int i = tid; i < 16 * KX_; i += 256) {
    int row = i / KX_, col = i - row * KX_;
    float v;
    if (col < 32)       v = states[((size_t)(r0 + row) * T_ + t) * DS_ + col];
    else if (col < 40)  v = actions[((size_t)(r0 + row) * T_ + t) * DA_ + (col - 32)];
    else if (col < 552) v = sH[row * 512 + (col - 40)];
    else                v = 0.0f;
    sA[i] = f2bf(v);
  }
  __syncthreads();
#pragma unroll 1
  for (int tn = wave; tn < 65; tn += 8) {
    if (tn + 8 < 65) prefetchB(wkvT + (size_t)(tn + 8) * 16 * KX_, KX_);
    v8f cz = {};
    v8f c = wmma_k(sA, KX_, wkvT + (size_t)tn * 16 * KX_, KX_, 18, cz);
    int col = tn * 16 + nn;
    float bb = (col < 1032) ? b_kv[col] : 0.0f;
#pragma unroll
    for (int g = 0; g < 8; ++g)
      sF[(g + 8 * half) * 1040 + col] = c[g] + bb;
  }
  __syncthreads();

  // ---- Phase 3a: v=tanh, lr=sigmoid, k=softmax per head ----
  for (int i = tid; i < 16 * 512; i += 256) {
    int row = i >> 9, j = i & 511;
    sF[row * 1040 + 512 + j] = fast_tanh(sF[row * 1040 + 512 + j]);
  }
  for (int i = tid; i < 16 * 8; i += 256) {
    int row = i >> 3, h = i & 7;
    sF[row * 1040 + 1024 + h] = fast_sigmoid(sF[row * 1040 + 1024 + h]);
  }
  if (tid < 128) {
    int row = tid >> 3, h = tid & 7;
    float* kp = &sF[row * 1040 + h * 64];
    float mx = kp[0];
    for (int j = 1; j < 64; ++j) mx = fmaxf(mx, kp[j]);
    float s = 0.0f;
    for (int j = 0; j < 64; ++j) { float e = __expf(kp[j] - mx); kp[j] = e; s += e; }
    float inv = __builtin_amdgcn_rcpf(s);
    for (int j = 0; j < 64; ++j) kp[j] *= inv;
  }
  __syncthreads();

  // ---- Phase 3b: q = softmax(LN(data@W_in+b_in)@W_q + b_q) ----
  for (int i = tid; i < 16 * 64; i += 256) {
    int row = i >> 6, col = i & 63;
    float v = 0.0f;
    if (col < 32)      v = states[((size_t)(r0 + row) * T_ + t) * DS_ + col];
    else if (col < 40) v = actions[((size_t)(r0 + row) * T_ + t) * DA_ + (col - 32)];
    sA[i] = f2bf(v);
  }
  __syncthreads();
#pragma unroll 1
  for (int tn = wave; tn < 32; tn += 8) {
    v8f cz = {};
    v8f c = wmma_k(sA, 64, winT + (size_t)tn * 16 * 64, 64, 2, cz);
    int col = tn * 16 + nn;
    float bb = b_in[col];
#pragma unroll
    for (int g = 0; g < 8; ++g)
      sH[(g + 8 * half) * 512 + col] = c[g] + bb;
  }
  __syncthreads();
  // LayerNorm(sH) -> bf16 sA
  {
    int row = tid >> 4, sub = tid & 15;
    float s = 0.0f, ss = 0.0f;
    for (int j = sub; j < 512; j += 16) { float x = sH[row * 512 + j]; s += x; ss += x * x; }
    sRed[row * 16 + sub] = s; sRed[256 + row * 16 + sub] = ss;
  }
  __syncthreads();
  if (tid < 16) {
    float s = 0.0f, ss = 0.0f;
    for (int j = 0; j < 16; ++j) { s += sRed[tid * 16 + j]; ss += sRed[256 + tid * 16 + j]; }
    float m = s * (1.0f / 512.0f), v = ss * (1.0f / 512.0f) - m * m;
    sMV[tid] = m; sMV[16 + tid] = __builtin_amdgcn_rsqf(v + 1e-5f);
  }
  __syncthreads();
  for (int i = tid; i < 16 * 512; i += 256) {
    int row = i >> 9, j = i & 511;
    sA[i] = f2bf((sH[i] - sMV[row]) * sMV[16 + row] * ln_in_g[j] + ln_in_b[j]);
  }
  __syncthreads();
#pragma unroll 1
  for (int tn = wave; tn < 32; tn += 8) {
    if (tn + 8 < 32) prefetchB(wqT + (size_t)(tn + 8) * 16 * 512, 512);
    v8f cz = {};
    v8f c = wmma_k(sA, 512, wqT + (size_t)tn * 16 * 512, 512, 16, cz);
    int col = tn * 16 + nn;
    float bb = b_q[col];
#pragma unroll
    for (int g = 0; g < 8; ++g)
      sH[(g + 8 * half) * 512 + col] = c[g] + bb;
  }
  __syncthreads();
  if (tid < 128) {  // q softmax per (row, head)
    int row = tid >> 3, h = tid & 7;
    float* qp = &sH[row * 512 + h * 64];
    float mx = qp[0];
    for (int j = 1; j < 64; ++j) mx = fmaxf(mx, qp[j]);
    float s = 0.0f;
    for (int j = 0; j < 64; ++j) { float e = __expf(qp[j] - mx); qp[j] = e; s += e; }
    float inv = __builtin_amdgcn_rcpf(s);
    for (int j = 0; j < 64; ++j) qp[j] *= inv;
  }
  __syncthreads();

  // ---- Phase 3c: fused W2 += lr * v (x) k  and  out = W2 @ q  (L2-resident) ----
  float* outb = (float*)sA;  // 16x512 f32 alias (sA free here)
#pragma unroll 1
  for (int it = 0; it < 32; ++it) {
    int tt = it * 256 + tid;
    int row = tt >> 9;
    int rem = tt & 511;
    int h = rem >> 6, vi = rem & 63;
    float lr = sF[row * 1040 + 1024 + h];
    float vv = sF[row * 1040 + 512 + h * 64 + vi];
    const float* kp = &sF[row * 1040 + h * 64];
    const float* qp = &sH[row * 512 + h * 64];
    float* wp = W2g + (((size_t)(r0 + row) * H_ + h) * HD_ + vi) * HD_;
    float lv = lr * vv, a = 0.0f;
#pragma unroll 4
    for (int kk = 0; kk < 64; ++kk) {
      float w = wp[kk] + lv * kp[kk];
      wp[kk] = w;
      a += w * qp[kk];
    }
    outb[row * 512 + rem] = a;
  }
  __syncthreads();

  // ---- Phase 4: attn_out = out @ W_out + b_out  -> sH ("o") ----
  for (int i = tid; i < 16 * 512; i += 256) sY[i] = f2bf(outb[i]);
  __syncthreads();
#pragma unroll 1
  for (int tn = wave; tn < 32; tn += 8) {
    if (tn + 8 < 32) prefetchB(woutT + (size_t)(tn + 8) * 16 * 512, 512);
    v8f cz = {};
    v8f c = wmma_k(sY, 512, woutT + (size_t)tn * 16 * 512, 512, 16, cz);
    int col = tn * 16 + nn;
    float bb = b_out[col];
#pragma unroll
    for (int g = 0; g < 8; ++g)
      sH[(g + 8 * half) * 512 + col] = c[g] + bb;
  }
  __syncthreads();

  // ---- Phase 5: y = LN(o); out = relu(y@W_ff1+b1)@W_ff2 + b2 + o ----
  {
    int row = tid >> 4, sub = tid & 15;
    float s = 0.0f, ss = 0.0f;
    for (int j = sub; j < 512; j += 16) { float x = sH[row * 512 + j]; s += x; ss += x * x; }
    sRed[row * 16 + sub] = s; sRed[256 + row * 16 + sub] = ss;
  }
  __syncthreads();
  if (tid < 16) {
    float s = 0.0f, ss = 0.0f;
    for (int j = 0; j < 16; ++j) { s += sRed[tid * 16 + j]; ss += sRed[256 + tid * 16 + j]; }
    float m = s * (1.0f / 512.0f), v = ss * (1.0f / 512.0f) - m * m;
    sMV[tid] = m; sMV[16 + tid] = __builtin_amdgcn_rsqf(v + 1e-5f);
  }
  __syncthreads();
  for (int i = tid; i < 16 * 512; i += 256) {
    int row = i >> 9, j = i & 511;
    sY[i] = f2bf((sH[i] - sMV[row]) * sMV[16 + row] * ln_ff_g[j] + ln_ff_b[j]);
  }
  __syncthreads();

  v8f cff[4];
#pragma unroll
  for (int ti = 0; ti < 4; ++ti) cff[ti] = {};
#pragma unroll 1
  for (int p = 0; p < 2; ++p) {      // FF=2048 processed in two 1024-wide halves
#pragma unroll 1
    for (int tn = wave; tn < 64; tn += 8) {  // ff1 half -> sF f32 (with relu)
      if (tn + 8 < 64) prefetchB(wff1T + (size_t)(p * 64 + tn + 8) * 16 * 512, 512);
      v8f cz = {};
      v8f c = wmma_k(sY, 512, wff1T + (size_t)(p * 64 + tn) * 16 * 512, 512, 16, cz);
      int col = tn * 16 + nn;
      float bb = b_ff1[p * 1024 + col];
#pragma unroll
      for (int g = 0; g < 8; ++g)
        sF[(g + 8 * half) * 1040 + col] = fmaxf(c[g] + bb, 0.0f);
    }
    __syncthreads();
    for (int i = tid; i < 16 * 1024; i += 256) {
      int row = i >> 10, j = i & 1023;
      sA[i] = f2bf(sF[row * 1040 + j]);
    }
    __syncthreads();
#pragma unroll
    for (int ti = 0; ti < 4; ++ti) {  // ff2 partial accumulation across halves
      int tn = wave + ti * 8;
      cff[ti] = wmma_k(sA, 1024, wff2T + (size_t)tn * 16 * FF_ + p * 1024, FF_, 32, cff[ti]);
    }
    __syncthreads();
  }
#pragma unroll
  for (int ti = 0; ti < 4; ++ti) {    // epilogue: +b_ff2 + residual o; carry h <- out
    int tn = wave + ti * 8;
    int col = tn * 16 + nn;
    float bb = b_ff2[col];
#pragma unroll
    for (int g = 0; g < 8; ++g) {
      int row = g + 8 * half;
      float o = cff[ti][g] + bb + sH[row * 512 + col];
      sF[row * 1040 + col] = o;
      hState[(size_t)(r0 + row) * 512 + col] = o;
    }
  }
  __syncthreads();

  // ---- Phase 6: pred = out @ W_dec + b_dec ----
  for (int i = tid; i < 16 * 512; i += 256)
    sY[i] = f2bf(sF[(i >> 9) * 1040 + (i & 511)]);
  __syncthreads();
  if (wave < 2) {
    int tn = wave;
    v8f cz = {};
    v8f c = wmma_k(sY, 512, wdecT + (size_t)tn * 16 * 512, 512, 16, cz);
    int col = tn * 16 + nn;
    float bb = b_dec[col];
#pragma unroll
    for (int g = 0; g < 8; ++g) {
      int row = g + 8 * half;
      outp[((size_t)(r0 + row) * T_ + t) * DS_ + col] = c[g] + bb;
    }
  }
}

// ---------------------------------------------------------------------------
extern "C" void kernel_launch(void* const* d_in, const int* in_sizes, int n_in,
                              void* d_out, int out_size, void* d_ws, size_t ws_size,
                              hipStream_t stream) {
  (void)in_sizes; (void)n_in; (void)out_size; (void)ws_size;
  const float* states  = (const float*)d_in[0];
  const float* actions = (const float*)d_in[1];
  const float* tsteps  = (const float*)d_in[2];
  const float* W_ode   = (const float*)d_in[3];
  const float* b_ode   = (const float*)d_in[4];
  const float* W_in    = (const float*)d_in[5];
  const float* b_in    = (const float*)d_in[6];
  const float* ln_in_g = (const float*)d_in[7];
  const float* ln_in_b = (const float*)d_in[8];
  const float* W_q     = (const float*)d_in[9];
  const float* b_q     = (const float*)d_in[10];
  const float* W_kv    = (const float*)d_in[11];
  const float* b_kv    = (const float*)d_in[12];
  const float* W_out   = (const float*)d_in[13];
  const float* b_out   = (const float*)d_in[14];
  const float* ln_ff_g = (const float*)d_in[15];
  const float* ln_ff_b = (const float*)d_in[16];
  const float* W_ff1   = (const float*)d_in[17];
  const float* b_ff1   = (const float*)d_in[18];
  const float* W_ff2   = (const float*)d_in[19];
  const float* b_ff2   = (const float*)d_in[20];
  const float* W_dec   = (const float*)d_in[21];
  const float* b_dec   = (const float*)d_in[22];

  char* ws = (char*)d_ws;
  float*  hState = (float*)(ws + OFF_H);
  float*  W2g    = (float*)(ws + OFF_W2);
  bf16_t* wodeT  = (bf16_t*)(ws + OFF_ODE);
  bf16_t* winT   = (bf16_t*)(ws + OFF_IN);
  bf16_t* wqT    = (bf16_t*)(ws + OFF_Q);
  bf16_t* wkvT   = (bf16_t*)(ws + OFF_KV);
  bf16_t* woutT  = (bf16_t*)(ws + OFF_OUT);
  bf16_t* wff1T  = (bf16_t*)(ws + OFF_FF1);
  bf16_t* wff2T  = (bf16_t*)(ws + OFF_FF2);
  bf16_t* wdecT  = (bf16_t*)(ws + OFF_DEC);

  // state init (deterministic, re-done every call)
  {
    int n1 = N_ * D_;
    zero_f32<<<(n1 + 255) / 256, 256, 0, stream>>>(hState, n1);
    int n2 = N_ * H_ * HD_ * HD_;
    zero_f32<<<(n2 + 255) / 256, 256, 0, stream>>>(W2g, n2);
  }
  // weight transpose + bf16 conversion
  auto cvt = [&](const float* src, bf16_t* dst, int K, int Nc, int Kp, int Np) {
    int total = Kp * Np;
    convert_wT<<<(total + 255) / 256, 256, 0, stream>>>(src, dst, K, Nc, Kp, Np);
  };
  cvt(W_ode, wodeT, 512, 512, 512, 512);
  cvt(W_in,  winT,  40,  512, KIN_, 512);
  cvt(W_q,   wqT,   512, 512, 512, 512);
  cvt(W_kv,  wkvT,  552, 1032, KX_, NKV_);
  cvt(W_out, woutT, 512, 512, 512, 512);
  cvt(W_ff1, wff1T, 512, 2048, 512, 2048);
  cvt(W_ff2, wff2T, 2048, 512, 2048, 512);
  cvt(W_dec, wdecT, 512, 32, 512, 32);

  // 64 sequential recurrent steps (dependency enforced by stream order)
  for (int t = 0; t < T_; ++t) {
    step_kernel<<<N_ / 16, 256, 0, stream>>>(
        states, actions, tsteps,
        b_ode, b_in, ln_in_g, ln_in_b, b_q, b_kv, b_out,
        ln_ff_g, ln_ff_b, b_ff1, b_ff2, b_dec,
        wodeT, winT, wqT, wkvT, woutT, wff1T, wff2T, wdecT,
        hState, W2g, (float*)d_out, t);
  }
}
